// sLSTMLayer_29875792511771
// MI455X (gfx1250) — matrix operand order repaired and checked by reading, not compiled, not run on hardware
//
#include <hip/hip_runtime.h>

// ---------------------------------------------------------------------------
// sLSTM layer for MI455X (gfx1250, wave32, WMMA, TDM).
//   Phase 1: LayerNorm(x) -> fp16 A matrix, laid out (s*B+b, d).
//   Phase 2: xW = A @ W_w^T + W_b via v_wmma_f32_16x16x32_f16; memory-bound
//            (~296MB HBM vs 137 GFLOP), so the work is pure data movement:
//            128-bit fragment loads, software-pipelined K loop, and a block
//            mapping where 8 waves share one N-group (B reuse via WGP$).
//   Phase 3: persistent recurrence (2048 steps). U (8MB fp16) is re-read from
//            the 192MB L2 once per step (the unavoidable traffic); h (32KB)
//            is staged into LDS per block per step via the Tensor Data Mover
//            (tensor_load_to_lds + s_wait_tensorcnt), A-fragments then come
//            from LDS. Device-wide sense barrier between steps (8 blocks).
// Workspace: 32MB(A16) + 8MB(W16) + 8MB(U16) + 256MB(xW) + state ~= 304MB.
// ---------------------------------------------------------------------------

typedef __attribute__((ext_vector_type(16))) _Float16 v16h;
typedef __attribute__((ext_vector_type(8)))  _Float16 v8h;
typedef __attribute__((ext_vector_type(8)))  float    v8f;

#if defined(__has_builtin)
#  if __has_builtin(__builtin_amdgcn_tensor_load_to_lds) && \
      __has_builtin(__builtin_amdgcn_s_wait_tensorcnt)
#    define HAVE_TDM 1
#  endif
#endif

#ifdef HAVE_TDM
typedef __attribute__((ext_vector_type(4))) unsigned u32x4;
typedef __attribute__((ext_vector_type(4))) int      i32x4;
typedef __attribute__((ext_vector_type(8))) int      i32x8;
#endif

static constexpr int Bb = 8;      // batch
static constexpr int Ss = 2048;   // seq len
static constexpr int Dd = 1024;   // hidden
static constexpr int G4 = 4096;   // 4*D
#define LN_EPS 1e-5f
#define N_EPS  1e-6f

// Load a 16-element f16 fragment as two contiguous 8-element (128-bit) runs.
// A and B layout (same order): runs at k+16h and k+16h+8.
__device__ __forceinline__ v16h load_frag16(const _Float16* p0, const _Float16* p1) {
  v8h r0 = *reinterpret_cast<const v8h*>(p0);
  v8h r1 = *reinterpret_cast<const v8h*>(p1);
  v16h a;
#pragma unroll
  for (int i = 0; i < 8; ++i) { a[i] = r0[i]; a[i + 8] = r1[i]; }
  return a;
}

__device__ __forceinline__ float sigmoidf_(float v) {
  return 1.0f / (1.0f + __expf(-v));
}

// ---------------------------------------------------------------------------
__global__ __launch_bounds__(256) void f32_to_f16_kernel(
    const float* __restrict__ src, _Float16* __restrict__ dst, int n) {
  int i = blockIdx.x * 256 + threadIdx.x;
  if (i < n) dst[i] = (_Float16)src[i];
}

// ---------------------------------------------------------------------------
// LayerNorm over D=1024; one block per (b,s) row; writes fp16 A[(s*B+b), d].
__global__ __launch_bounds__(256) void layernorm_f16_kernel(
    const float* __restrict__ x, const float* __restrict__ gamma,
    const float* __restrict__ beta, _Float16* __restrict__ A) {
  __shared__ float red[256];
  const int bs = blockIdx.x;            // = b*S + s (x row order)
  const int b  = bs >> 11;
  const int s  = bs & 2047;
  const float* xr = x + (size_t)bs * Dd;

  float lsum = 0.f;
  for (int i = threadIdx.x; i < Dd; i += 256) lsum += xr[i];
  red[threadIdx.x] = lsum; __syncthreads();
  for (int o = 128; o > 0; o >>= 1) {
    if (threadIdx.x < o) red[threadIdx.x] += red[threadIdx.x + o];
    __syncthreads();
  }
  const float mean = red[0] * (1.f / Dd);
  __syncthreads();

  float lvar = 0.f;
  for (int i = threadIdx.x; i < Dd; i += 256) {
    float dv = xr[i] - mean; lvar += dv * dv;
  }
  red[threadIdx.x] = lvar; __syncthreads();
  for (int o = 128; o > 0; o >>= 1) {
    if (threadIdx.x < o) red[threadIdx.x] += red[threadIdx.x + o];
    __syncthreads();
  }
  const float rstd = rsqrtf(red[0] * (1.f / Dd) + LN_EPS);

  _Float16* Ar = A + ((size_t)s * Bb + b) * Dd;
  for (int i = threadIdx.x; i < Dd; i += 256)
    Ar[i] = (_Float16)((xr[i] - mean) * rstd * gamma[i] + beta[i]);
}

// ---------------------------------------------------------------------------
__global__ __launch_bounds__(256) void init_state_kernel(
    _Float16* __restrict__ hst, float* __restrict__ cst,
    float* __restrict__ nst, unsigned* __restrict__ ctrl) {
  int i = blockIdx.x * 256 + threadIdx.x;
  if (i < 16 * Dd) hst[i] = (_Float16)0.f;   // rows 8..15 = zero M padding
  if (i < Bb * Dd) { cst[i] = 0.f; nst[i] = 1.f; }
  if (i < 2) ctrl[i] = 0u;
}

// ---------------------------------------------------------------------------
// xW GEMM: C(16384 x 4096) = A(16384 x 1024) * W^T + bias.
// Block = 8 waves sharing ONE 64-col N-group over 8 consecutive M-tiles, so
// all waves issue identical B addresses (WGP$ reuse of the 8MB weight).
// Register-double-buffered K loop: next iteration's fragments are in flight
// while the current four WMMAs execute.
__global__ __launch_bounds__(256) void xw_gemm_kernel(
    const _Float16* __restrict__ A, const _Float16* __restrict__ W,
    const float* __restrict__ bias, float* __restrict__ xW) {
  const int lane = threadIdx.x & 31;
  const int wave = threadIdx.x >> 5;
  const int ng   = blockIdx.x >> 7;                    // 0..63 N-group
  const int mt   = ((blockIdx.x & 127) << 3) + wave;   // 0..1023 M-tile
  const int n0   = ng << 6;
  const int half = lane >> 4;
  const int l16  = lane & 15;

  const _Float16* Arow = A + (size_t)(mt * 16 + l16) * Dd;
  const _Float16* W0 = W + (size_t)(n0 +  0 + l16) * Dd;
  const _Float16* W1 = W + (size_t)(n0 + 16 + l16) * Dd;
  const _Float16* W2 = W + (size_t)(n0 + 32 + l16) * Dd;
  const _Float16* W3 = W + (size_t)(n0 + 48 + l16) * Dd;

  v8f c0 = {}, c1 = {}, c2 = {}, c3 = {};
  // Prologue: fragments for k = 0.
  v16h a  = load_frag16(Arow + 16 * half, Arow + 16 * half + 8);
  v16h b0 = load_frag16(W0 + 16 * half, W0 + 16 * half + 8);
  v16h b1 = load_frag16(W1 + 16 * half, W1 + 16 * half + 8);
  v16h b2 = load_frag16(W2 + 16 * half, W2 + 16 * half + 8);
  v16h b3 = load_frag16(W3 + 16 * half, W3 + 16 * half + 8);
  for (int k = 0; k < Dd; k += 32) {
    const int kn = (k + 32 < Dd) ? (k + 32) : k;   // clamp (redundant last)
    __builtin_prefetch(Arow + k + 256, 0, 3);
    v16h an  = load_frag16(Arow + kn + 16 * half, Arow + kn + 16 * half + 8);
    v16h nb0 = load_frag16(W0 + kn + 16 * half, W0 + kn + 16 * half + 8);
    v16h nb1 = load_frag16(W1 + kn + 16 * half, W1 + kn + 16 * half + 8);
    v16h nb2 = load_frag16(W2 + kn + 16 * half, W2 + kn + 16 * half + 8);
    v16h nb3 = load_frag16(W3 + kn + 16 * half, W3 + kn + 16 * half + 8);
    c0 = __builtin_amdgcn_wmma_f32_16x16x32_f16(false, a, false, b0, (short)0, c0, false, false);
    c1 = __builtin_amdgcn_wmma_f32_16x16x32_f16(false, a, false, b1, (short)0, c1, false, false);
    c2 = __builtin_amdgcn_wmma_f32_16x16x32_f16(false, a, false, b2, (short)0, c2, false, false);
    c3 = __builtin_amdgcn_wmma_f32_16x16x32_f16(false, a, false, b3, (short)0, c3, false, false);
    a = an; b0 = nb0; b1 = nb1; b2 = nb2; b3 = nb3;
  }

  // C/D layout: VGPR v, lanes 0-15 -> M = v; lanes 16-31 -> M = v+8; N = l16.
  const int m0 = mt * 16 + 8 * half;
#pragma unroll
  for (int v = 0; v < 8; ++v) {
    float* o = xW + (size_t)(m0 + v) * G4 + n0 + l16;
    o[0]  = c0[v] + bias[n0 +  0 + l16];
    o[16] = c1[v] + bias[n0 + 16 + l16];
    o[32] = c2[v] + bias[n0 + 32 + l16];
    o[48] = c3[v] + bias[n0 + 48 + l16];
  }
}

// ---------------------------------------------------------------------------
// Device-wide sense-reversing barrier across NB co-resident blocks.
__device__ __forceinline__ void grid_sync(unsigned* cnt, unsigned* sense, unsigned nb) {
  __threadfence();
  __syncthreads();
  if (threadIdx.x == 0) {
    unsigned old = __atomic_load_n(sense, __ATOMIC_RELAXED);
    if (__atomic_add_fetch(cnt, 1u, __ATOMIC_ACQ_REL) == nb) {
      __atomic_store_n(cnt, 0u, __ATOMIC_RELAXED);
      __atomic_fetch_add(sense, 1u, __ATOMIC_RELEASE);
    } else {
      while (__atomic_load_n(sense, __ATOMIC_ACQUIRE) == old)
        __builtin_amdgcn_s_sleep(2);
    }
  }
  __syncthreads();
  (void)__atomic_load_n(sense, __ATOMIC_ACQUIRE);
}

// ---------------------------------------------------------------------------
// Persistent recurrence: grid = 8 blocks x 256 threads = 64 waves.
// Per step: stage h (16x1024 f16, 32KB) into LDS (TDM if available), then
// wave t computes the four gate tiles at N = g*1024 + t*16 with A-fragments
// from LDS and B-fragments (U columns) from L2, register-double-buffered.
__global__ __launch_bounds__(256) void slstm_scan_kernel(
    const float* __restrict__ x, const float* __restrict__ xW,
    const _Float16* __restrict__ U16, _Float16* __restrict__ hst,
    float* __restrict__ cst, float* __restrict__ nst,
    float* __restrict__ out, unsigned* __restrict__ ctrl) {
  __shared__ _Float16 hlds[16 * Dd];   // 32KB staging for h
  const int lane = threadIdx.x & 31;
  const int wave = threadIdx.x >> 5;
  const int t    = blockIdx.x * 8 + wave;   // d-tile 0..63
  const int half = lane >> 4;
  const int l16  = lane & 15;
  const int d    = t * 16 + l16;

  const _Float16* Arow = hlds + (size_t)l16 * Dd;           // A row = M = l16
  const _Float16* U0 = U16 + ((size_t)0 * Dd + d) * Dd;     // gate i column
  const _Float16* U1 = U16 + ((size_t)1 * Dd + d) * Dd;     // gate f column
  const _Float16* U2 = U16 + ((size_t)2 * Dd + d) * Dd;     // gate o column
  const _Float16* U3 = U16 + ((size_t)3 * Dd + d) * Dd;     // gate z column

#ifdef HAVE_TDM
  // Tensor DMA descriptor: 1-row tile of 4096 x 8-byte units (= 32KB of h).
  const unsigned ldsb = (unsigned)(unsigned long long)
      (__attribute__((address_space(3))) _Float16*)hlds;
  const unsigned long long ga = (unsigned long long)hst;
  u32x4 g0;
  g0[0] = 1u;                                   // count=1 (valid descriptor)
  g0[1] = ldsb;                                 // lds_addr (bytes)
  g0[2] = (unsigned)(ga & 0xffffffffu);         // global_addr[31:0]
  g0[3] = (unsigned)((ga >> 32) & 0x01ffffffu)  // global_addr[56:32]
          | 0x80000000u;                        // type = 2 ("image")
  i32x8 g1;
  g1[0] = 3 << 16;                 // data_size = 3 (8 bytes)
  g1[1] = (int)(0x1000u << 16);    // tensor_dim0[15:0] = 4096 in bits[31:16]
  g1[2] = 1 << 16;                 // tensor_dim0 hi = 0; tensor_dim1 = 1
  g1[3] = (int)(0x1000u << 16);    // tensor_dim1 hi = 0; tile_dim0 = 4096
  g1[4] = 1;                       // tile_dim1 = 1; tile_dim2 = 0
  g1[5] = 4096;                    // tensor_dim0_stride = 4096 units
  g1[6] = 0;
  g1[7] = 0;
  const i32x4 z4 = {0, 0, 0, 0};
#endif

  for (int s = 0; s < Ss; ++s) {
    // --- stage h -> LDS -------------------------------------------------
#ifdef HAVE_TDM
    if (threadIdx.x == 0) {
#  if __clang_major__ >= 23
      const i32x8 z8 = {0, 0, 0, 0, 0, 0, 0, 0};
      __builtin_amdgcn_tensor_load_to_lds(g0, g1, z4, z4, z8, 0);
#  else
      __builtin_amdgcn_tensor_load_to_lds(g0, g1, z4, z4, 0);
#  endif
      __builtin_amdgcn_s_wait_tensorcnt(0);
    }
    __syncthreads();
#else
    for (int i = threadIdx.x; i < (16 * Dd) / 8; i += 256)
      ((uint4*)hlds)[i] = ((const uint4*)hst)[i];
    __syncthreads();
#endif

    // --- h @ U^T for the 4 gate tiles (pipelined K loop) ---------------
    v8f c0 = {}, c1 = {}, c2 = {}, c3 = {};
    v16h b0 = load_frag16(U0 + 16 * half, U0 + 16 * half + 8);
    v16h b1 = load_frag16(U1 + 16 * half, U1 + 16 * half + 8);
    v16h b2 = load_frag16(U2 + 16 * half, U2 + 16 * half + 8);
    v16h b3 = load_frag16(U3 + 16 * half, U3 + 16 * half + 8);
    for (int k = 0; k < Dd; k += 32) {
      const int kn = (k + 32 < Dd) ? (k + 32) : k;
      v16h a   = load_frag16(Arow + k + 16 * half, Arow + k + 16 * half + 8);
      v16h nb0 = load_frag16(U0 + kn + 16 * half, U0 + kn + 16 * half + 8);
      v16h nb1 = load_frag16(U1 + kn + 16 * half, U1 + kn + 16 * half + 8);
      v16h nb2 = load_frag16(U2 + kn + 16 * half, U2 + kn + 16 * half + 8);
      v16h nb3 = load_frag16(U3 + kn + 16 * half, U3 + kn + 16 * half + 8);
      c0 = __builtin_amdgcn_wmma_f32_16x16x32_f16(false, a, false, b0, (short)0, c0, false, false);
      c1 = __builtin_amdgcn_wmma_f32_16x16x32_f16(false, a, false, b1, (short)0, c1, false, false);
      c2 = __builtin_amdgcn_wmma_f32_16x16x32_f16(false, a, false, b2, (short)0, c2, false, false);
      c3 = __builtin_amdgcn_wmma_f32_16x16x32_f16(false, a, false, b3, (short)0, c3, false, false);
      b0 = nb0; b1 = nb1; b2 = nb2; b3 = nb3;
    }

    // --- pointwise sLSTM update (lanes 0-15: VGPR v <-> batch row b=v) --
    if (half == 0) {
#pragma unroll
      for (int v = 0; v < 8; ++v) {
        const float* xr = xW + ((size_t)s * Bb + v) * G4 + d;
        float gi = c0[v] + xr[0 * Dd];
        float gf = c1[v] + xr[1 * Dd];
        float go = c2[v] + xr[2 * Dd];
        float gz = c3[v] + xr[3 * Dd];
        float it = __expf(gi);
        float ft = sigmoidf_(gf);
        float ot = sigmoidf_(go);
        float zt = tanhf(gz);
        size_t si = (size_t)v * Dd + d;
        float cv = ft * cst[si] + it * zt;
        float nv = ft * nst[si] + it;
        float hv = ot * (cv / (nv + N_EPS));
        cst[si] = cv;
        nst[si] = nv;
        hst[si] = (_Float16)hv;             // rows 8..15 stay zero (padding)
        size_t oi = ((size_t)v * Ss + s) * Dd + d;
        out[oi] = x[oi] + hv;               // y residual add
      }
    }
    grid_sync(ctrl, ctrl + 1, (unsigned)gridDim.x);
  }
}

// ---------------------------------------------------------------------------
extern "C" void kernel_launch(void* const* d_in, const int* in_sizes, int n_in,
                              void* d_out, int out_size, void* d_ws, size_t ws_size,
                              hipStream_t stream) {
  const float* x     = (const float*)d_in[0];
  const float* W_w   = (const float*)d_in[1];
  const float* W_b   = (const float*)d_in[2];
  const float* U_w   = (const float*)d_in[3];
  const float* gamma = (const float*)d_in[4];
  const float* beta  = (const float*)d_in[5];
  float* out = (float*)d_out;

  char* ws = (char*)d_ws;
  size_t off = 0;
  _Float16* h16A = (_Float16*)(ws + off); off += (size_t)Bb * Ss * Dd * 2;  // 32MB
  _Float16* Ww16 = (_Float16*)(ws + off); off += (size_t)G4 * Dd * 2;       //  8MB
  _Float16* Uw16 = (_Float16*)(ws + off); off += (size_t)G4 * Dd * 2;       //  8MB
  float*    xW   = (float*)(ws + off);    off += (size_t)Bb * Ss * G4 * 4;  // 256MB
  _Float16* hst  = (_Float16*)(ws + off); off += (size_t)16 * Dd * 2;
  float*    cst  = (float*)(ws + off);    off += (size_t)Bb * Dd * 4;
  float*    nst  = (float*)(ws + off);    off += (size_t)Bb * Dd * 4;
  unsigned* ctrl = (unsigned*)(ws + off);

  const int nw = G4 * Dd;  // 4,194,304 weights per matrix
  f32_to_f16_kernel<<<(nw + 255) / 256, 256, 0, stream>>>(W_w, Ww16, nw);
  f32_to_f16_kernel<<<(nw + 255) / 256, 256, 0, stream>>>(U_w, Uw16, nw);
  layernorm_f16_kernel<<<Bb * Ss, 256, 0, stream>>>(x, gamma, beta, h16A);
  init_state_kernel<<<64, 256, 0, stream>>>(hst, cst, nst, ctrl);
  xw_gemm_kernel<<<8192, 256, 0, stream>>>(h16A, Ww16, W_b, xW);
  slstm_scan_kernel<<<8, 256, 0, stream>>>(x, xW, Uw16, hst, cst, nst, out, ctrl);
}
